// DenseCINConv_80676665688179
// MI455X (gfx1250) — matrix operand order repaired and checked by reading, not compiled, not run on hardware
//
#include <hip/hip_runtime.h>
#include <hip/hip_bf16.h>

typedef __attribute__((ext_vector_type(16))) _Float16 v16h;
typedef __attribute__((ext_vector_type(8)))  float    v8f;

#define BN_EPS 1e-5f

// ---------------------------------------------------------------------------
// Elementwise helpers
// ---------------------------------------------------------------------------
__global__ void copy4_kernel(float4* __restrict__ dst, const float4* __restrict__ src, int n4) {
    int t = blockIdx.x * blockDim.x + threadIdx.x;
    if (t < n4) dst[t] = src[t];
}

__global__ void zero_kernel(float* __restrict__ p, int n) {
    int t = blockIdx.x * blockDim.x + threadIdx.x;
    if (t < n) p[t] = 0.0f;
}

// scatter: dst[tgt[e]] += src[srcIdx[e]]  (64 floats per edge, 4 per thread)
__global__ void scatter_add_kernel(float* __restrict__ dst, const float* __restrict__ src,
                                   const int* __restrict__ idx, int n_e) {
    long long t = (long long)blockIdx.x * blockDim.x + threadIdx.x;
    long long total = (long long)n_e * 16;
    if (t >= total) return;
    int e = (int)(t >> 4);
    int c = ((int)t & 15) * 4;
    int s = idx[e];            // idx row 0 : source rows
    int g = idx[n_e + e];      // idx row 1 : target rows
    float4 v = *(const float4*)(src + (size_t)s * 64 + c);
    float* d = dst + (size_t)g * 64 + c;
    atomicAdd(d + 0, v.x); atomicAdd(d + 1, v.y);
    atomicAdd(d + 2, v.z); atomicAdd(d + 3, v.w);
}

// Pack W (K x 64, f32 row-major) -> Wt (64 x K, f16 row-major).  Each WMMA
// B-fragment then reads two 8-half (16B) runs, 16 halves apart, per lane.
__global__ void pack_wt_kernel(const float* __restrict__ W, _Float16* __restrict__ out, int K) {
    int t = blockIdx.x * blockDim.x + threadIdx.x;
    int total = K * 64;
    if (t < total) {
        int n = t / K;
        int k = t - n * K;
        out[t] = (_Float16)W[(size_t)k * 64 + n];
    }
}

// stats[n]=sum, stats[64+n]=sumsq  ->  scale/shift of affine BN (+ gamma/beta)
__global__ void finalize_bn_kernel(const float* __restrict__ stats,
                                   const float* __restrict__ g, const float* __restrict__ bt,
                                   float invN, float* __restrict__ scale, float* __restrict__ shift) {
    int n = threadIdx.x;              // 64 threads
    float mu  = stats[n] * invN;
    float var = stats[64 + n] * invN - mu * mu;
    float rstd = rsqrtf(var + BN_EPS);
    float sc = g[n] * rstd;
    scale[n] = sc;
    shift[n] = bt[n] - mu * sc;
}

// y = relu(sc[c]*x + sh[c]) with independent in/out row strides
__global__ void bn_relu_kernel(const float* __restrict__ in, int inLd,
                               float* __restrict__ out, int outLd,
                               const float* __restrict__ sc, const float* __restrict__ sh,
                               int total) {
    for (int t = blockIdx.x * blockDim.x + threadIdx.x; t < total;
         t += gridDim.x * blockDim.x) {
        int row = t >> 6, c = t & 63;
        float v = in[(size_t)row * inLd + c];
        out[(size_t)row * outLd + c] = fmaxf(0.0f, sc[c] * v + sh[c]);
    }
}

// ---------------------------------------------------------------------------
// WMMA GEMM: out[:, colOff:colOff+64] = act(X) @ W + bias ; accumulate BN stats
//   X: N x K row-major f32; Wt: pre-packed f16 transpose (64 x K).
//   TRANSFORM: act(x) = relu(inScale[k]*x + inShift[k]) (upstream BN fused
//   into the A-matrix load); otherwise identity.
//   One wave -> 16 rows x 64 cols (4 accumulators), v_wmma_f32_16x16x32_f16.
// ---------------------------------------------------------------------------
template <bool TRANSFORM>
__global__ __launch_bounds__(256)
void gemm_bn_stats_kernel(const float* __restrict__ X, int N, int K,
                          const float* __restrict__ inScale, const float* __restrict__ inShift,
                          const _Float16* __restrict__ Wt, const float* __restrict__ bias,
                          float* __restrict__ out, int ldOut, int colOff,
                          float* __restrict__ stats) {
    const int w    = threadIdx.x >> 5;          // wave in block (8 waves)
    const int lane = threadIdx.x & 31;
    const int nlo  = lane & 15;
    const int hi   = lane >> 4;                 // lane half
    const int row0 = (blockIdx.x * 8 + w) * 16; // wave's 16-row tile
    if (row0 >= N) return;                      // wave-uniform exit

    int rowC = row0 + nlo;                      // A-row for this lane (clamped)
    if (rowC >= N) rowC = N - 1;
    const float* xr = X + (size_t)rowC * K;

    v8f acc[4] = {};

    for (int k0 = 0; k0 < K; k0 += 32) {
        // ---- A fragment (16x32 f16): lane half picks K {0..7,16..23} / {8..15,24..31}
        const int kb0 = k0 + hi * 8;
        const int kb1 = kb0 + 16;
        float4 p0 = *(const float4*)(xr + kb0);
        float4 p1 = *(const float4*)(xr + kb0 + 4);
        float4 p2 = *(const float4*)(xr + kb1);
        float4 p3 = *(const float4*)(xr + kb1 + 4);
        float va[16] = {p0.x, p0.y, p0.z, p0.w, p1.x, p1.y, p1.z, p1.w,
                        p2.x, p2.y, p2.z, p2.w, p3.x, p3.y, p3.z, p3.w};
        v16h a;
        if (TRANSFORM) {
            float4 s0 = *(const float4*)(inScale + kb0);
            float4 s1 = *(const float4*)(inScale + kb0 + 4);
            float4 s2 = *(const float4*)(inScale + kb1);
            float4 s3 = *(const float4*)(inScale + kb1 + 4);
            float4 h0 = *(const float4*)(inShift + kb0);
            float4 h1 = *(const float4*)(inShift + kb0 + 4);
            float4 h2 = *(const float4*)(inShift + kb1);
            float4 h3 = *(const float4*)(inShift + kb1 + 4);
            float sa[16] = {s0.x, s0.y, s0.z, s0.w, s1.x, s1.y, s1.z, s1.w,
                            s2.x, s2.y, s2.z, s2.w, s3.x, s3.y, s3.z, s3.w};
            float ha[16] = {h0.x, h0.y, h0.z, h0.w, h1.x, h1.y, h1.z, h1.w,
                            h2.x, h2.y, h2.z, h2.w, h3.x, h3.y, h3.z, h3.w};
#pragma unroll
            for (int j = 0; j < 16; ++j)
                a[j] = (_Float16)fmaxf(0.0f, sa[j] * va[j] + ha[j]);
        } else {
#pragma unroll
            for (int j = 0; j < 16; ++j)
                a[j] = (_Float16)va[j];
        }
        // ---- 4 column tiles of B: two 16B f16 loads per fragment, K {0..7,16..23} / {8..15,24..31}
#pragma unroll
        for (int t = 0; t < 4; ++t) {
            const _Float16* wb = Wt + (size_t)(t * 16 + nlo) * K + k0 + hi * 8;
            typedef __attribute__((ext_vector_type(8))) _Float16 v8h_t;
            v8h_t blo = *(const v8h_t*)(wb);
            v8h_t bhi = *(const v8h_t*)(wb + 16);
            v16h bf = __builtin_shufflevector(blo, bhi, 0,1,2,3,4,5,6,7,8,9,10,11,12,13,14,15);
            acc[t] = __builtin_amdgcn_wmma_f32_16x16x32_f16(
                false, a, false, bf, (short)0, acc[t], false, false);
        }
    }

    // ---- epilogue: +bias, store, BN stats (sum & sumsq per output feature).
    //      Fast straight-line path for full tiles (wave-uniform test); only the
    //      single possibly-partial tail tile takes the guarded path.
    const bool fullTile = (row0 + 16 <= N);
    float* op0 = out + (size_t)(row0 + 8 * hi) * ldOut + colOff;
#pragma unroll
    for (int t = 0; t < 4; ++t) {
        int n = t * 16 + nlo;
        float bv = bias[n];
        float s1 = 0.0f, s2 = 0.0f;
        if (fullTile) {
#pragma unroll
            for (int r = 0; r < 8; ++r) {
                float v = acc[t][r] + bv;
                op0[(size_t)r * ldOut + n] = v;
                s1 += v; s2 += v * v;
            }
        } else {
#pragma unroll
            for (int r = 0; r < 8; ++r) {
                int row = row0 + r + 8 * hi;    // C/D layout: m = r + 8*hi
                float v = acc[t][r] + bv;
                if (row < N) {
                    out[(size_t)row * ldOut + colOff + n] = v;
                    s1 += v; s2 += v * v;
                }
            }
        }
        s1 += __shfl_xor(s1, 16);
        s2 += __shfl_xor(s2, 16);
        if (lane < 16) {
            atomicAdd(&stats[n], s1);
            atomicAdd(&stats[64 + n], s2);
        }
    }
}

// ---------------------------------------------------------------------------
// Host orchestration
// ---------------------------------------------------------------------------
struct Mlp  { const float *W1,*b1,*g1,*bt1,*W2,*b2,*g2,*bt2; };
struct Comb { const float *Wc,*bc,*gc,*btc; };

extern "C" void kernel_launch(void* const* d_in, const int* in_sizes, int n_in,
                              void* d_out, int out_size, void* d_ws, size_t ws_size,
                              hipStream_t stream) {
    constexpr int N0 = 100000, N1 = 200000, N2 = 75000;

    auto F = [&](int i) { return (const float*)d_in[i]; };

    const float* x0 = F(0);
    const float* x1 = F(1);
    const float* x2 = F(2);

    // params leaf order: detect dict-insertion vs sorted-key flattening.
    bool sortedLeaves = (n_in > 4 && in_sizes[4] == 4096); // W1,W2,... => sorted
    int ip = 3;
    auto takeMlp = [&]() -> Mlp {
        Mlp m;
        if (!sortedLeaves) { // W1,b1,g1,bt1,W2,b2,g2,bt2
            m.W1=F(ip++); m.b1=F(ip++); m.g1=F(ip++); m.bt1=F(ip++);
            m.W2=F(ip++); m.b2=F(ip++); m.g2=F(ip++); m.bt2=F(ip++);
        } else {             // W1,W2,b1,b2,bt1,bt2,g1,g2
            m.W1=F(ip++); m.W2=F(ip++); m.b1=F(ip++); m.b2=F(ip++);
            m.bt1=F(ip++); m.bt2=F(ip++); m.g1=F(ip++); m.g2=F(ip++);
        }
        return m;
    };
    auto takeComb = [&]() -> Comb {
        Comb c;
        if (!sortedLeaves) { c.Wc=F(ip++); c.bc=F(ip++); c.gc=F(ip++); c.btc=F(ip++); }
        else               { c.Wc=F(ip++); c.bc=F(ip++); c.btc=F(ip++); c.gc=F(ip++); }
        return c;
    };

    Mlp d0_up, d0_cob; Comb d0_c;
    Mlp d1_up, d1_down, d1_bnd, d1_cob; Comb d1_c;
    Mlp d2_down, d2_bnd; Comb d2_c;
    if (!sortedLeaves) { // insertion order: d0{up,cob,comb} d1{up,down,bnd,cob,comb} d2{down,bnd,comb}
        d0_up=takeMlp(); d0_cob=takeMlp(); d0_c=takeComb();
        d1_up=takeMlp(); d1_down=takeMlp(); d1_bnd=takeMlp(); d1_cob=takeMlp(); d1_c=takeComb();
        d2_down=takeMlp(); d2_bnd=takeMlp(); d2_c=takeComb();
    } else {             // sorted keys: d0{cob,comb,up} d1{bnd,cob,comb,down,up} d2{bnd,comb,down}
        d0_cob=takeMlp(); d0_c=takeComb(); d0_up=takeMlp();
        d1_bnd=takeMlp(); d1_cob=takeMlp(); d1_c=takeComb(); d1_down=takeMlp(); d1_up=takeMlp();
        d2_bnd=takeMlp(); d2_c=takeComb(); d2_down=takeMlp();
    }
    const int* up0   = (const int*)d_in[ip]; int up0_ne   = in_sizes[ip]/2; ip++;
    const int* cob0  = (const int*)d_in[ip]; int cob0_ne  = in_sizes[ip]/2; ip++;
    const int* up1   = (const int*)d_in[ip]; int up1_ne   = in_sizes[ip]/2; ip++;
    const int* down1 = (const int*)d_in[ip]; int down1_ne = in_sizes[ip]/2; ip++;
    const int* bnd1  = (const int*)d_in[ip]; int bnd1_ne  = in_sizes[ip]/2; ip++;
    const int* cob1  = (const int*)d_in[ip]; int cob1_ne  = in_sizes[ip]/2; ip++;
    const int* down2 = (const int*)d_in[ip]; int down2_ne = in_sizes[ip]/2; ip++;
    const int* bnd2  = (const int*)d_in[ip]; int bnd2_ne  = in_sizes[ip]/2; ip++;

    // ---- workspace layout (floats)
    float* ws = (float*)d_ws;
    size_t o = 0;
    float* tmpA   = ws + o; o += (size_t)N1 * 64;    // agg buffer / combine pre-BN
    float* tmpB   = ws + o; o += (size_t)N1 * 64;    // GEMM1 pre-BN output
    float* concat = ws + o; o += (size_t)N1 * 256;   // branch concat, PRE-BN (reused per dim)
    float* sml    = ws + o; o += 2048;
    _Float16* wpack = (_Float16*)(ws + o); o += 8192; // 16384 halves (max K=256)
    float* stats1    = sml;        float* stats2    = sml + 128;
    float* scale1    = sml + 256;  float* shift1    = sml + 320;
    float* combScale = sml + 384;  float* combShift = sml + 640;  // up to 256 each

    auto launchGemm = [&](const float* X, int N, int K,
                          const float* isc, const float* ish,
                          const float* W, const float* b,
                          float* out, int ldOut, int colOff, float* st) {
        int total = K * 64;
        pack_wt_kernel<<<(total + 255) / 256, 256, 0, stream>>>(W, wpack, K);
        dim3 grid((N + 127) / 128);
        if (isc)
            gemm_bn_stats_kernel<true><<<grid, 256, 0, stream>>>(
                X, N, K, isc, ish, wpack, b, out, ldOut, colOff, st);
        else
            gemm_bn_stats_kernel<false><<<grid, 256, 0, stream>>>(
                X, N, K, nullptr, nullptr, wpack, b, out, ldOut, colOff, st);
    };

    // ---- one branch: agg + residual -> Linear/BN/ReLU -> Linear (BN2 deferred
    //      into the combine GEMM's A-load via combScale/combShift)
    auto runBranch = [&](const float* x_tgt, const float* x_src,
                         const int* idx, int n_e, int N, const Mlp& p,
                         int ldC, int colOff) {
        int n4 = N * 16;
        copy4_kernel<<<(n4 + 255) / 256, 256, 0, stream>>>((float4*)tmpA,
                                                           (const float4*)x_tgt, n4);
        long long tot = (long long)n_e * 16;
        scatter_add_kernel<<<(unsigned)((tot + 255) / 256), 256, 0, stream>>>(
            tmpA, x_src, idx, n_e);
        zero_kernel<<<1, 256, 0, stream>>>(sml, 256);           // stats1+stats2
        launchGemm(tmpA, N, 64, nullptr, nullptr, p.W1, p.b1, tmpB, 64, 0, stats1);
        finalize_bn_kernel<<<1, 64, 0, stream>>>(stats1, p.g1, p.bt1, 1.0f / N,
                                                 scale1, shift1);
        launchGemm(tmpB, N, 64, scale1, shift1, p.W2, p.b2, concat, ldC, colOff, stats2);
        finalize_bn_kernel<<<1, 64, 0, stream>>>(stats2, p.g2, p.bt2, 1.0f / N,
                                                 combScale + colOff, combShift + colOff);
    };

    auto runCombine = [&](int N, int numAdj, const Comb& c, float* outPtr) {
        zero_kernel<<<1, 256, 0, stream>>>(stats1, 128);
        launchGemm(concat, N, numAdj * 64, combScale, combShift, c.Wc, c.bc,
                   tmpA, 64, 0, stats1);
        finalize_bn_kernel<<<1, 64, 0, stream>>>(stats1, c.gc, c.btc, 1.0f / N,
                                                 scale1, shift1);
        int total = N * 64;
        bn_relu_kernel<<<(total + 255) / 256, 256, 0, stream>>>(tmpA, 64, outPtr, 64,
                                                                scale1, shift1, total);
    };

    float* out = (float*)d_out;

    // dim 0: [up, cob] -> combine
    runBranch(x0, x0, up0,  up0_ne,  N0, d0_up,  128, 0);
    runBranch(x0, x1, cob0, cob0_ne, N0, d0_cob, 128, 64);
    runCombine(N0, 2, d0_c, out);

    // dim 1: [up, down, bnd, cob] -> combine
    runBranch(x1, x1, up1,   up1_ne,   N1, d1_up,   256, 0);
    runBranch(x1, x1, down1, down1_ne, N1, d1_down, 256, 64);
    runBranch(x1, x0, bnd1,  bnd1_ne,  N1, d1_bnd,  256, 128);
    runBranch(x1, x2, cob1,  cob1_ne,  N1, d1_cob,  256, 192);
    runCombine(N1, 4, d1_c, out + (size_t)N0 * 64);

    // dim 2: [down, bnd] -> combine
    runBranch(x2, x2, down2, down2_ne, N2, d2_down, 128, 0);
    runBranch(x2, x1, bnd2,  bnd2_ne,  N2, d2_bnd,  128, 64);
    runCombine(N2, 2, d2_c, out + (size_t)(N0 + N1) * 64);
}
